// SymmetrizationMLP_11982958756740
// MI455X (gfx1250) — matrix operand order repaired and checked by reading, not compiled, not run on hardware
//
#include <hip/hip_runtime.h>

typedef __attribute__((ext_vector_type(16))) _Float16 v16h;
typedef __attribute__((ext_vector_type(8)))  _Float16 v8h;
typedef __attribute__((ext_vector_type(8)))  float    v8f;

#define B_   1024
#define NPT  1024
#define KP   6
#define G_   720
#define H1_  64
#define H2_  32
#define C_   40
#define IN_  18
#define APAD 24   // aperm row stride: 18 data + [18]=1.0 (bias) + [19..23]=0

// f32 accumulator rows -> f16 with relu applied in packed f16
// (cvt results are canonical -> plain v_pk_max_num_f16).
__device__ __forceinline__ v8h relu_cvt8(v8f a) {
  v8h hv;
#pragma unroll
  for (int r = 0; r < 8; ++r) hv[r] = (_Float16)a[r];
  v8h z = {};
  return __builtin_elementwise_max(hv, z);
}

__global__ __launch_bounds__(256)
void symm_mlp_kernel(const float* __restrict__ x,
                     const float* __restrict__ W1, const float* __restrict__ b1,
                     const float* __restrict__ W2, const float* __restrict__ b2,
                     const float* __restrict__ W3, const float* __restrict__ b3,
                     const int*   __restrict__ idx, const int* __restrict__ perms,
                     float* __restrict__ out) {
  __shared__ float pts_s[IN_];
  __shared__ unsigned char perms_s[G_ * KP];
  __shared__ float hsum_s[H2_];            // column sums of relu(H2)
  __shared__ _Float16 wt1_s[H1_ * 32];     // W1^T (+bias row), [col][K], K=0..31
  __shared__ _Float16 wt2_s[H2_ * 64];     // W2^T, [col][K], K=0..63
  __shared__ _Float16 aperm_s[G_ * APAD];  // permuted inputs + bias column, f16
  __shared__ _Float16 h1_s[8][16 * H1_];   // per-wave scratch, layer-1 tile

  const int b     = blockIdx.x;
  const int tid   = threadIdx.x;
  const int lane  = tid & 31;
  const int wave  = tid >> 5;
  const int col16 = lane & 15;
  const int kgrp  = lane >> 4;

  // ---- stage permutations (bytes) and the 6 gathered points ----
  for (int i = tid; i < G_ * KP; i += 256)
    perms_s[i] = (unsigned char)perms[i];
  if (tid < IN_) {
    int k = tid / 3, d = tid - 3 * k;
    int p = idx[b * KP + k];
    pts_s[tid] = x[((size_t)b * NPT + p) * 3 + d];
  }
  if (tid < H2_) hsum_s[tid] = 0.0f;

  // ---- stage W1^T (+b1 as K=18) and W2^T into LDS, coalesced, f16 ----
#pragma unroll
  for (int j = 0; j < (H1_ * 32) / 256; ++j) {
    int i   = tid + j * 256;
    int col = i & 63, K = i >> 6;        // consecutive tid -> consecutive col
    float v = (K < IN_) ? W1[K * H1_ + col] : (K == IN_ ? b1[col] : 0.0f);
    wt1_s[col * 32 + K] = (_Float16)v;
  }
#pragma unroll
  for (int j = 0; j < (H2_ * 64) / 256; ++j) {
    int i   = tid + j * 256;
    int col = i & 31, K = i >> 5;        // consecutive tid -> consecutive col
    wt2_s[col * 64 + K] = (_Float16)W2[K * H2_ + col];
  }

  // ---- layer-2 bias as persistent C fragments (bias depends only on N) ----
  v8f cb2[2];
#pragma unroll
  for (int n = 0; n < 2; ++n) {
    float bv = b2[n * 16 + col16];
#pragma unroll
    for (int r = 0; r < 8; ++r) cb2[n][r] = bv;
  }

  __syncthreads();

  // ---- weight fragments from LDS: W1 in 16B+8B+8B runs, W2 in contiguous 32B runs ----
  v16h w1f[4];
#pragma unroll
  for (int n = 0; n < 4; ++n) {
    const _Float16* p = &wt1_s[(n * 16 + col16) * 32];
#pragma unroll
    for (int e = 0; e < 16; ++e) w1f[n][e] = p[e < 8 ? 8 * kgrp + e : (e < 12 ? 16 + 4 * kgrp + (e - 8) : 24 + 4 * kgrp + (e - 12))];
  }
  v16h w2f[2][2];
#pragma unroll
  for (int n = 0; n < 2; ++n)
#pragma unroll
    for (int c = 0; c < 2; ++c) {
      const _Float16* p = &wt2_s[(n * 16 + col16) * 64 + c * 32 + kgrp * 16];
#pragma unroll
      for (int e = 0; e < 16; ++e) w2f[n][c][e] = p[e];
    }

  // ---- materialize the 720x18 permuted matrix once (f16, padded rows) ----
  for (int g = tid; g < G_; g += 256) {
    const unsigned char* pr = &perms_s[g * KP];
    _Float16* dst = &aperm_s[g * APAD];
#pragma unroll
    for (int k = 0; k < KP; ++k) {
      int p = pr[k];
      dst[3 * k + 0] = (_Float16)pts_s[p * 3 + 0];
      dst[3 * k + 1] = (_Float16)pts_s[p * 3 + 1];
      dst[3 * k + 2] = (_Float16)pts_s[p * 3 + 2];
    }
    dst[18] = (_Float16)1.0f;   // constant-1 column -> multiplies b1 row of W1
#pragma unroll
    for (int k = 19; k < APAD; ++k) dst[k] = (_Float16)0.0f;
  }

  __syncthreads();

  // running sums for relu(H2) columns: sum(x) and sum(|x|) -> max at the end
  float s0 = 0.0f, sa0 = 0.0f, s1 = 0.0f, sa1 = 0.0f;

  // per-lane A1 base pointer, strength-reduced across the tile loop
  const _Float16* arow = aperm_s + (size_t)(wave * 16 + col16) * APAD;

  // ---- 45 M-tiles of 16 rows, round-robin over the 8 waves ----
  for (int t = wave; t < 45; t += 8, arow += 8 * 16 * APAD) {
    // A1 (16x32 f16): e0..7 -> b128 run at 8*kgrp; e8..11 -> b64 run at 16+4*kgrp
    v16h a1;
#pragma unroll
    for (int e = 0; e < 8; ++e) a1[e] = arow[8 * kgrp + e];
#pragma unroll
    for (int e = 0; e < 4; ++e) a1[8 + e] = arow[16 + 4 * kgrp + e];
#pragma unroll
    for (int e = 12; e < 16; ++e) a1[e] = (_Float16)0.0f;

    // ---- layer 1: 18(+bias)->64; 4 WMMAs, then one packed epilogue ----
    v8f acc1[4];
#pragma unroll
    for (int n = 0; n < 4; ++n) {
      v8f z = {};
      acc1[n] = __builtin_amdgcn_wmma_f32_16x16x32_f16(false, a1, false, w1f[n],
                                                       (short)0, z, false, false);
    }
#pragma unroll
    for (int n = 0; n < 4; ++n) {
      v8h hv = relu_cvt8(acc1[n]);
#pragma unroll
      for (int r = 0; r < 8; ++r)
        h1_s[wave][(r + 8 * kgrp) * H1_ + n * 16 + col16] = hv[r];
    }

    // ---- layer 2: 64->32; bias via C operand; WMMAs grouped ----
    v16h a2c0, a2c1;
#pragma unroll
    for (int e = 0; e < 16; ++e) {
      int Kb = e + 16 * kgrp;
      a2c0[e] = h1_s[wave][col16 * H1_ + Kb];
      a2c1[e] = h1_s[wave][col16 * H1_ + 32 + Kb];
    }
    v8f acc2[2];
#pragma unroll
    for (int n = 0; n < 2; ++n)
      acc2[n] = __builtin_amdgcn_wmma_f32_16x16x32_f16(false, a2c0, false, w2f[n][0],
                                                       (short)0, cb2[n], false, false);
#pragma unroll
    for (int n = 0; n < 2; ++n)
      acc2[n] = __builtin_amdgcn_wmma_f32_16x16x32_f16(false, a2c1, false, w2f[n][1],
                                                       (short)0, acc2[n], false, false);

    // ---- layer 3 commuted with mean: sum(max(x,0)) = (sum x + sum|x|)/2 ----
#pragma unroll
    for (int r = 0; r < 8; ++r) { s0 += acc2[0][r]; sa0 += fabsf(acc2[0][r]); }
#pragma unroll
    for (int r = 0; r < 8; ++r) { s1 += acc2[1][r]; sa1 += fabsf(acc2[1][r]); }
  }

  // ---- fold lane halves, one atomic per (n, column) per wave ----
  {
    float hs0 = 0.5f * (s0 + sa0);
    float hs1 = 0.5f * (s1 + sa1);
    hs0 += __shfl_xor(hs0, 16, 32);
    hs1 += __shfl_xor(hs1, 16, 32);
    if (kgrp == 0) {
      atomicAdd(&hsum_s[col16], hs0);
      atomicAdd(&hsum_s[16 + col16], hs1);
    }
  }

  __syncthreads();

  // ---- final 32x40 matvec once per block: mean(H2) @ W3 + b3 ----
  if (tid < C_) {
    float acc = 0.0f;
#pragma unroll
    for (int k = 0; k < H2_; ++k) acc += hsum_s[k] * W3[k * C_ + tid];
    out[(size_t)b * C_ + tid] = acc * (1.0f / (float)G_) + b3[tid];
  }
}

extern "C" void kernel_launch(void* const* d_in, const int* in_sizes, int n_in,
                              void* d_out, int out_size, void* d_ws, size_t ws_size,
                              hipStream_t stream) {
  const float* x  = (const float*)d_in[0];
  const float* W1 = (const float*)d_in[1];
  const float* b1 = (const float*)d_in[2];
  const float* W2 = (const float*)d_in[3];
  const float* b2 = (const float*)d_in[4];
  const float* W3 = (const float*)d_in[5];
  const float* b3 = (const float*)d_in[6];
  const int*   idx   = (const int*)d_in[7];
  const int*   perms = (const int*)d_in[8];
  float* out = (float*)d_out;

  symm_mlp_kernel<<<B_, 256, 0, stream>>>(x, W1, b1, W2, b2, W3, b3, idx, perms, out);
  (void)in_sizes; (void)n_in; (void)out_size; (void)d_ws; (void)ws_size;
}
